// GraphMixer_7980049236593
// MI455X (gfx1250) — matrix operand order repaired and checked by reading, not compiled, not run on hardware
//
#include <hip/hip_runtime.h>

typedef __attribute__((ext_vector_type(16))) _Float16 v16h;
typedef __attribute__((ext_vector_type(8)))  float    v8f;

#define NNODES   50000
#define NEDGES   500000
#define NDIM     256
#define EDIM     64
#define NHEAD    4
#define CDIM     64
#define HC       256
#define NGRAPH   64
#define NEG_SLOPE 0.2f
#define ENC_NEG_INF 0x007FFFFFu   /* encf(-inf) */

// Monotone float<->uint mapping so unsigned atomicMax == float max
__device__ __forceinline__ unsigned encf(float f) {
    unsigned u = __float_as_uint(f);
    return (u & 0x80000000u) ? ~u : (u | 0x80000000u);
}
__device__ __forceinline__ float decf(unsigned u) {
    unsigned b = (u & 0x80000000u) ? (u ^ 0x80000000u) : ~u;
    return __uint_as_float(b);
}

// ---------------- init: zero accumulators, set segment-max to -inf ----------
__global__ void k_init(unsigned* __restrict__ menc, float* __restrict__ denom,
                       float* __restrict__ hsum, float* __restrict__ gsum,
                       float* __restrict__ gcnt) {
    int i = blockIdx.x * blockDim.x + threadIdx.x;
    if (i < NNODES * NHEAD) { menc[i] = ENC_NEG_INF; denom[i] = 0.0f; }
    if (i < NNODES * HC)    hsum[i] = 0.0f;
    if (i < NGRAPH * HC)    gsum[i] = 0.0f;
    if (i < NGRAPH)         gcnt[i] = 0.0f;
}

// ---------------- x = node_feats @ W via v_wmma_f32_16x16x32_f16 ------------
// One wave computes TWO 16x16 tiles (register blocking: one A fragment feeds
// two WMMAs per K-step); 8 waves/block cover all 16 column tiles of a 16-row
// strip. f32 loads converted to f16 in-register, f32 accumulation.
__global__ __launch_bounds__(256) void k_gemm_x(const float* __restrict__ NF,
                                                const float* __restrict__ Wm,
                                                float* __restrict__ xb) {
    const int wave = threadIdx.x >> 5;
    const int lane = threadIdx.x & 31;
    const int l15  = lane & 15;
    const int m0   = blockIdx.x * 16;
    const int m    = m0 + l15;
    const int kbaseA = (lane < 16) ? 0 : 8;    // A-fragment K origin (ISA layout)
    const int kbaseB = (lane < 16) ? 0 : 16;   // unused; B fragments also use kbaseA
    const float* rowA = NF + (size_t)m * NDIM;

    const int na = wave * 16 + l15;            // column of tile 0
    const int nb = (wave + 8) * 16 + l15;      // column of tile 1
    v8f acc0 = {}, acc1 = {};

    for (int k0 = 0; k0 < NDIM; k0 += 32) {
        // prefetch next A block (streamed, row-major)
        if (k0 + 32 < NDIM) __builtin_prefetch(rowA + k0 + 32, 0, 3);
        // A fragment: lane holds row m, K = kbaseA + {0..7} and kbaseA+16+{0..7}
        const float4* pa0 = (const float4*)(rowA + k0 + kbaseA);
        const float4* pa1 = (const float4*)(rowA + k0 + kbaseA + 16);
        float4 p0 = pa0[0], p1 = pa0[1];
        float4 p2 = pa1[0], p3 = pa1[1];
        v16h a;
        a[0]=(_Float16)p0.x;  a[1]=(_Float16)p0.y;  a[2]=(_Float16)p0.z;  a[3]=(_Float16)p0.w;
        a[4]=(_Float16)p1.x;  a[5]=(_Float16)p1.y;  a[6]=(_Float16)p1.z;  a[7]=(_Float16)p1.w;
        a[8]=(_Float16)p2.x;  a[9]=(_Float16)p2.y;  a[10]=(_Float16)p2.z; a[11]=(_Float16)p2.w;
        a[12]=(_Float16)p3.x; a[13]=(_Float16)p3.y; a[14]=(_Float16)p3.z; a[15]=(_Float16)p3.w;
        // B fragments: lane holds column n, K = kbaseA + {0..7} and kbaseA+16+{0..7}
        v16h b0, b1;
        #pragma unroll
        for (int j = 0; j < 16; ++j) {
            const float* wrow = Wm + (size_t)(k0 + kbaseA + (j & 7) + ((j >> 3) << 4)) * HC;
            b0[j] = (_Float16)wrow[na];
            b1[j] = (_Float16)wrow[nb];
        }
        acc0 = __builtin_amdgcn_wmma_f32_16x16x32_f16(
            false, a, false, b0, (short)0, acc0, false, false);
        acc1 = __builtin_amdgcn_wmma_f32_16x16x32_f16(
            false, a, false, b1, (short)0, acc1, false, false);
    }
    // D layout: VGPR r -> row (r) for lanes 0-15, row (r+8) for lanes 16-31
    const int mb = m0 + ((lane < 16) ? 0 : 8);
    #pragma unroll
    for (int r = 0; r < 8; ++r) {
        xb[(size_t)(mb + r) * HC + na] = acc0[r];
        xb[(size_t)(mb + r) * HC + nb] = acc1[r];
    }
}

// ---------------- per-node attention scalars a_src, a_dst -------------------
__global__ void k_att_nodes(const float* __restrict__ xb,
                            const float* __restrict__ att_src,
                            const float* __restrict__ att_dst,
                            float* __restrict__ as_, float* __restrict__ ad_) {
    int tid = blockIdx.x * blockDim.x + threadIdx.x;
    if (tid >= NNODES * NHEAD) return;
    int h = tid & 3;
    int n = tid >> 2;
    const float* xr = xb + (size_t)n * HC + h * CDIM;
    const float* s  = att_src + h * CDIM;
    const float* d  = att_dst + h * CDIM;
    float accS = 0.f, accD = 0.f;
    #pragma unroll 8
    for (int c = 0; c < CDIM; ++c) {
        float xv = xr[c];
        accS += xv * s[c];
        accD += xv * d[c];
    }
    as_[tid] = accS;
    ad_[tid] = accD;
}

// ---------------- V[ed,h] = sum_c lin_edge_W[ed, h*C+c] * att_edge[h,c] -----
__global__ void k_edge_vec(const float* __restrict__ leW,
                           const float* __restrict__ att_edge,
                           float* __restrict__ Vb) {
    int t = threadIdx.x;              // 256 threads = EDIM*NHEAD
    int ed = t >> 2, h = t & 3;
    float acc = 0.f;
    #pragma unroll 8
    for (int c = 0; c < CDIM; ++c)
        acc += leW[(size_t)ed * HC + h * CDIM + c] * att_edge[h * CDIM + c];
    Vb[ed * NHEAD + h] = acc;
}

// ---------------- per-edge logits + LeakyReLU + segment max -----------------
__global__ __launch_bounds__(256) void k_alpha(const float* __restrict__ ea,
                       const int* __restrict__ src, const int* __restrict__ dst,
                       const float* __restrict__ as_, const float* __restrict__ ad_,
                       const float* __restrict__ Vb,
                       float* __restrict__ alpha, unsigned* __restrict__ menc) {
    __shared__ float Vs[EDIM * NHEAD];
    Vs[threadIdx.x] = Vb[threadIdx.x];
    __syncthreads();
    int e = blockIdx.x * blockDim.x + threadIdx.x;
    if (e >= NEDGES) return;
    int s = src[e], d = dst[e];
    const float4 av = *(const float4*)(as_ + (size_t)s * NHEAD);  // 16B aligned
    const float4 dv = *(const float4*)(ad_ + (size_t)d * NHEAD);
    float acc[NHEAD] = { av.x + dv.x, av.y + dv.y, av.z + dv.z, av.w + dv.w };
    const float* er = ea + (size_t)e * EDIM;
    #pragma unroll 4
    for (int ed = 0; ed < EDIM; ++ed) {
        float v = er[ed];
        #pragma unroll
        for (int h = 0; h < NHEAD; ++h)
            acc[h] += v * Vs[ed * NHEAD + h];
    }
    float4 outv;
    #pragma unroll
    for (int h = 0; h < NHEAD; ++h) {
        float a0 = acc[h];
        a0 = (a0 > 0.f) ? a0 : NEG_SLOPE * a0;
        ((float*)&outv)[h] = a0;
        atomicMax(&menc[d * NHEAD + h], encf(a0));
    }
    *(float4*)(alpha + (size_t)e * NHEAD) = outv;
}

// ---------------- exp(alpha - max) and segment denominator ------------------
__global__ void k_expdenom(const int* __restrict__ dst,
                           float* __restrict__ alpha,
                           const unsigned* __restrict__ menc,
                           float* __restrict__ denom) {
    int e = blockIdx.x * blockDim.x + threadIdx.x;
    if (e >= NEDGES) return;
    int d = dst[e];
    const uint4  mv = *(const uint4*)(menc + (size_t)d * NHEAD);
    float4 av = *(float4*)(alpha + (size_t)e * NHEAD);
    float4 ev;
    ev.x = expf(av.x - decf(mv.x));
    ev.y = expf(av.y - decf(mv.y));
    ev.z = expf(av.z - decf(mv.z));
    ev.w = expf(av.w - decf(mv.w));
    *(float4*)(alpha + (size_t)e * NHEAD) = ev;      // now holds exp(alpha - m)
    atomicAdd(&denom[d * NHEAD + 0], ev.x);
    atomicAdd(&denom[d * NHEAD + 1], ev.y);
    atomicAdd(&denom[d * NHEAD + 2], ev.z);
    atomicAdd(&denom[d * NHEAD + 3], ev.w);
}

// ---------------- message scatter: hsum[dst] += attn * x[src] ---------------
// 64 threads per edge, float4 per thread (256 channels). x and hsum both live
// in the 192MB L2, so gathers/atomics mostly stay on-chip.
__global__ void k_scatter(const int* __restrict__ src, const int* __restrict__ dst,
                          const float* __restrict__ xb,
                          const float* __restrict__ alpha,
                          const float* __restrict__ denom,
                          float* __restrict__ hsum) {
    int t = blockIdx.x * blockDim.x + threadIdx.x;
    int e = t >> 6;
    if (e >= NEDGES) return;
    int q  = t & 63;
    int hc = q * 4;
    int h  = hc >> 6;                 // CDIM = 64
    int s = src[e], d = dst[e];
    float attn = alpha[(size_t)e * NHEAD + h] / (denom[d * NHEAD + h] + 1e-16f);
    const float4 xv = *(const float4*)(xb + (size_t)s * HC + hc);
    float* o = hsum + (size_t)d * HC + hc;
    atomicAdd(o + 0, xv.x * attn);
    atomicAdd(o + 1, xv.y * attn);
    atomicAdd(o + 2, xv.z * attn);
    atomicAdd(o + 3, xv.w * attn);
}

// ---------------- bias + ReLU + per-graph sum & count -----------------------
__global__ void k_pool(const float* __restrict__ hsum, const float* __restrict__ bias,
                       const int* __restrict__ batch,
                       float* __restrict__ gsum, float* __restrict__ gcnt) {
    int n  = blockIdx.x;
    int hc = threadIdx.x;
    float v = hsum[(size_t)n * HC + hc] + bias[hc];
    v = (v > 0.f) ? v : 0.f;
    int g = batch[n];
    atomicAdd(&gsum[g * HC + hc], v);
    if (hc == 0) atomicAdd(&gcnt[g], 1.0f);
}

// ---------------- final: mean pool + hyper projection -----------------------
__global__ void k_final(const float* __restrict__ gsum, const float* __restrict__ gcnt,
                        const float* __restrict__ hyperW, const float* __restrict__ hyperb,
                        float* __restrict__ out) {
    __shared__ float red[HC];
    int g = blockIdx.x, t = threadIdx.x;
    float cnt = gcnt[g];
    cnt = (cnt < 1.f) ? 1.f : cnt;
    red[t] = (gsum[g * HC + t] / cnt) * hyperW[t];
    __syncthreads();
    #pragma unroll
    for (int sft = HC / 2; sft > 0; sft >>= 1) {
        if (t < sft) red[t] += red[t + sft];
        __syncthreads();
    }
    if (t == 0) out[g] = red[0] + hyperb[0];
}

extern "C" void kernel_launch(void* const* d_in, const int* in_sizes, int n_in,
                              void* d_out, int out_size, void* d_ws, size_t ws_size,
                              hipStream_t stream) {
    const float* node_feats = (const float*)d_in[0];
    const float* edge_attr  = (const float*)d_in[1];
    const int*   edge_index = (const int*)d_in[2];
    const int*   batch      = (const int*)d_in[3];
    const float* Wm         = (const float*)d_in[4];
    const float* att_src    = (const float*)d_in[5];
    const float* att_dst    = (const float*)d_in[6];
    const float* lin_edge_W = (const float*)d_in[7];
    const float* att_edge   = (const float*)d_in[8];
    const float* bias       = (const float*)d_in[9];
    const float* hyper_W    = (const float*)d_in[10];
    const float* hyper_b    = (const float*)d_in[11];
    const int* src = edge_index;
    const int* dst = edge_index + NEDGES;
    float* out = (float*)d_out;

    // workspace carve-up (all fully (re)initialized or fully overwritten per call)
    float* xb    = (float*)d_ws;                       // N*HC
    float* as_   = xb + (size_t)NNODES * HC;           // N*H
    float* ad_   = as_ + (size_t)NNODES * NHEAD;       // N*H
    float* Vb    = ad_ + (size_t)NNODES * NHEAD;       // EDIM*H
    float* alpha = Vb + EDIM * NHEAD;                  // E*H
    unsigned* menc = (unsigned*)(alpha + (size_t)NEDGES * NHEAD); // N*H
    float* denom = (float*)(menc + (size_t)NNODES * NHEAD);       // N*H
    float* hsum  = denom + (size_t)NNODES * NHEAD;     // N*HC
    float* gsum  = hsum + (size_t)NNODES * HC;         // G*HC
    float* gcnt  = gsum + NGRAPH * HC;                 // G

    k_init<<<(NNODES * HC + 255) / 256, 256, 0, stream>>>(menc, denom, hsum, gsum, gcnt);
    k_gemm_x<<<NNODES / 16, 256, 0, stream>>>(node_feats, Wm, xb);
    k_att_nodes<<<(NNODES * NHEAD + 255) / 256, 256, 0, stream>>>(xb, att_src, att_dst, as_, ad_);
    k_edge_vec<<<1, 256, 0, stream>>>(lin_edge_W, att_edge, Vb);
    k_alpha<<<(NEDGES + 255) / 256, 256, 0, stream>>>(edge_attr, src, dst, as_, ad_, Vb, alpha, menc);
    k_expdenom<<<(NEDGES + 255) / 256, 256, 0, stream>>>(dst, alpha, menc, denom);
    k_scatter<<<(NEDGES * 64) / 256, 256, 0, stream>>>(src, dst, xb, alpha, denom, hsum);
    k_pool<<<NNODES, 256, 0, stream>>>(hsum, bias, batch, gsum, gcnt);
    k_final<<<NGRAPH, 256, 0, stream>>>(gsum, gcnt, hyper_W, hyper_b, out);
}
